// StableCNNLSTM_40269613367632
// MI455X (gfx1250) — matrix operand order repaired and checked by reading, not compiled, not run on hardware
//
#include <hip/hip_runtime.h>
#include <math.h>

// ---------------------------------------------------------------------------
// StableCNNLSTM forward for MI455X (gfx1250, wave32, WMMA 16x16x32 f16).
//
// Roofline: dominant GEMM (img proj, 10.7 GFLOP over 84MB f32) is HBM-bound
// at 23.3 TB/s -> f16 WMMA with f32 accumulation. A-tiles are DMA'd to LDS
// with the Tensor Data Mover (double-buffered, TENSORcnt-synchronized) and
// converted f32->f16 once per block instead of once per wave. Sequential
// scans (BiLSTM x4, ConvLSTM) are persistent-workgroup WMMA kernels with
// pre-swizzled f16 weight fragments and h-state in LDS. Attention is
// flash-style, one wave per (b, head, 16-query tile).
// ---------------------------------------------------------------------------

typedef __attribute__((ext_vector_type(16))) _Float16     v16h;
typedef __attribute__((ext_vector_type(8)))  float        v8f;
typedef __attribute__((ext_vector_type(4)))  unsigned int v4u;
typedef __attribute__((ext_vector_type(8)))  int          v8i;
typedef __attribute__((ext_vector_type(4)))  int          v4i;

#define WMMA(a, b, c) \
  __builtin_amdgcn_wmma_f32_16x16x32_f16(false, (a), false, (b), (short)0, (c), false, false)

#define BB   32
#define TT   512
#define BT   16384   // BB*TT

__device__ __forceinline__ float sigf(float x)  { return 1.f / (1.f + __expf(-x)); }
__device__ __forceinline__ float geluf(float x) { return 0.5f * x * (1.f + erff(x * 0.70710678118654752f)); }

__device__ __forceinline__ v16h load_v16h(const _Float16* p) {
  v16h r;
#pragma unroll
  for (int i = 0; i < 16; ++i) r[i] = p[i];
  return r;
}

// ---------------------------------------------------------------------------
// Tensor Data Mover: DMA a 16-row x 64-col f32 tile (row stride K elements)
// from global memory into LDS. D# layout per CDNA5 ISA ch.8 (08_async_tensor):
//   group0: count=1 | lds_addr | global_addr[56:0] | type=2
//   group1: data_size=2(4B), tensor_dim0=64, tensor_dim1=16,
//           tile_dim0=64, tile_dim1=16, tensor_dim0_stride=K
// 2D tensor -> groups 2/3 zero. Tracked with TENSORcnt.
// ---------------------------------------------------------------------------
__device__ __forceinline__ void tdm_load_tile16x64(const float* gsrc, unsigned lds_off, int K) {
  unsigned long long ga = (unsigned long long)(__SIZE_TYPE__)gsrc;
  v4u g0;
  g0[0] = 1u;                                               // count=1 (valid user D#)
  g0[1] = lds_off;                                          // lds_addr (bytes)
  g0[2] = (unsigned)(ga & 0xffffffffu);                     // global_addr[31:0]
  g0[3] = (unsigned)((ga >> 32) & 0x1ffffffu) | (2u << 30); // global_addr[56:32] | type=2
  v8i g1;
  g1[0] = (int)(2u << 16);   // wg_mask=0, data_size=2 (4 bytes), no pad/iter
  g1[1] = (int)(64u << 16);  // tensor_dim0[15:0] = 64
  g1[2] = (int)(16u << 16);  // tensor_dim0[31:16]=0, tensor_dim1[15:0] = 16
  g1[3] = (int)(64u << 16);  // tensor_dim1[31:16]=0, tile_dim0 = 64
  g1[4] = (int)16u;          // tile_dim1 = 16, tile_dim2 = 0
  g1[5] = (int)K;            // tensor_dim0_stride[31:0] = K elements
  g1[6] = 0;                 // stride hi / dim1_stride lo
  g1[7] = 0;
  v4i z4 = {0, 0, 0, 0};
#if defined(__clang_major__) && (__clang_major__ >= 23)
  v8i z8 = {0, 0, 0, 0, 0, 0, 0, 0};
  __builtin_amdgcn_tensor_load_to_lds(g0, g1, z4, z4, z8, 0);
#else
  __builtin_amdgcn_tensor_load_to_lds(g0, g1, z4, z4, 0);
#endif
}

// ===========================================================================
// Weight pre-swizzle: pack Whh (4 x [256,64]) and pw_W ([512,256]) into WMMA
// B-fragment order (f16). B-frag lane map (16-bit B, KxN=32x16):
//   col N = lane&15 ; halves h=0..7 -> K = (lane<16 ? 0 : 8) + h, h=8..15 -> K = (lane<16 ? 0 : 8) + 8 + h
// ===========================================================================
__global__ void pack_frags(const float* __restrict__ whh_ef, const float* __restrict__ whh_eb,
                           const float* __restrict__ whh_ff, const float* __restrict__ whh_fb,
                           const float* __restrict__ pw,
                           _Float16* __restrict__ whh16, _Float16* __restrict__ pw16) {
  int e = blockIdx.x * blockDim.x + threadIdx.x;
  if (e < 65536) {  // 4 dirs * 2 ktiles * 16 coltiles * 32 lanes * 16 halves
    int hh = e & 15, lane = (e >> 4) & 31, n = (e >> 9) & 15, kk = (e >> 13) & 1, dir = (e >> 14) & 3;
    const float* W = (dir == 0) ? whh_ef : (dir == 1) ? whh_eb : (dir == 2) ? whh_ff : whh_fb;
    int col = lane & 15;
    int k   = kk * 32 + ((lane < 16) ? 0 : 8) + ((hh < 8) ? hh : hh + 8);
    whh16[e] = (_Float16)W[(n * 16 + col) * 64 + k];
  } else if (e < 65536 + 131072) {  // 8 ktiles * 32 coltiles * 32 lanes * 16 halves
    int e2 = e - 65536;
    int hh = e2 & 15, lane = (e2 >> 4) & 31, n = (e2 >> 9) & 31, kk = (e2 >> 14) & 7;
    int col = lane & 15;
    int k   = kk * 32 + ((lane < 16) ? 0 : 8) + ((hh < 8) ? hh : hh + 8);
    pw16[e2] = (_Float16)pw[(n * 16 + col) * 256 + k];
  }
}

// ===========================================================================
// Generic WMMA GEMM: out[M,N](+ldo/colofs) = act(A[M,K] @ W[N,K]^T + bias)
// mode 0: bias only; mode 1: bias + LayerNorm(N); mode 2: bias + GELU + LN(N)
// grid.x = M/16 ; blockDim = (N/16)*32 ; K multiple of 64.
// A-tile path: TDM DMA (global->LDS f32, double buffered) -> one cooperative
// f32->f16 convert -> per-wave A-frag ds reads (kills the 16x redundant
// global loads + cvt of the naive scheme).
// A-frag lane map (16-bit A, 16x32): row M = lane&15;
//   halves 0..7  -> K = base + h      (base = lane<16 ? 0 : 16)
//   halves 8..15 -> K = base + h   (contiguous order used here; B matches)
// C-frag: row = r + (lane<16?0:8), col = lane&15.
// ===========================================================================
__global__ void gemm_wmma(const float* __restrict__ A, int K,
                          const float* __restrict__ W, const float* __restrict__ bias,
                          const float* __restrict__ g, const float* __restrict__ be,
                          float* __restrict__ out, int ldo, int colofs, int mode) {
  __shared__ float    tile[16 * 256];
  __shared__ float    a32[2][16 * 64];   // TDM destination (double buffer)
  __shared__ _Float16 a16[16 * 64];      // converted A chunk shared by all waves
  const int tid   = threadIdx.x;
  const int nthr  = blockDim.x;
  const int lane  = tid & 31;
  const int w     = tid >> 5;
  const int nw    = nthr >> 5;
  const int N     = nw * 16;
  const int rt    = blockIdx.x;
  const int half8 = (lane < 16) ? 0 : 8;
  const int b16   = (lane < 16) ? 0 : 16;
  const int mrow  = lane & 15;
  const int col   = w * 16 + mrow;

  const float* abase = A + (size_t)rt * 16 * K;
  const float* wrow  = W + (size_t)col * K;
  const int nc = K >> 6;  // 64-wide K chunks

  if (w == 0) tdm_load_tile16x64(abase, (unsigned)(__SIZE_TYPE__)&a32[0][0], K);

  v8f acc = {0.f, 0.f, 0.f, 0.f, 0.f, 0.f, 0.f, 0.f};
  for (int c = 0; c < nc; ++c) {
    if (w == 0) {
      if (c + 1 < nc) {  // prefetch next chunk via TDM, then ensure current done
        tdm_load_tile16x64(abase + (c + 1) * 64, (unsigned)(__SIZE_TYPE__)&a32[(c + 1) & 1][0], K);
        __builtin_amdgcn_s_wait_tensorcnt(1);
      } else {
        __builtin_amdgcn_s_wait_tensorcnt(0);
      }
    }
    __syncthreads();  // a32[c&1] ready; prev-iter a16 readers done
    const float* ac = &a32[c & 1][0];
    for (int i = tid; i < 16 * 64; i += nthr) a16[i] = (_Float16)ac[i];
    __syncthreads();

#pragma unroll
    for (int half = 0; half < 2; ++half) {  // two 32-wide WMMA K-steps per chunk
      const int kloc = half * 32;
      v16h af, bf;
#pragma unroll
      for (int hh = 0; hh < 8; ++hh) {
        af[hh]     = a16[mrow * 64 + kloc + b16 + hh];
        af[hh + 8] = a16[mrow * 64 + kloc + b16 + 8 + hh];
      }
      const float* wk = wrow + c * 64 + kloc + b16;
#pragma unroll
      for (int hh = 0; hh < 16; ++hh) bf[hh] = (_Float16)wk[hh];
      acc = WMMA(af, bf, acc);
    }
  }

  float bv = bias[col];
#pragma unroll
  for (int r = 0; r < 8; ++r) acc[r] += bv;

  if (mode == 0) {
#pragma unroll
    for (int r = 0; r < 8; ++r)
      out[(size_t)(rt * 16 + r + half8) * ldo + colofs + col] = acc[r];
    return;
  }

#pragma unroll
  for (int r = 0; r < 8; ++r) tile[(r + half8) * N + col] = acc[r];
  __syncthreads();

  for (int row = w; row < 16; row += nw) {
    float s = 0.f, ss = 0.f;
    for (int c = lane; c < N; c += 32) {
      float v = tile[row * N + c];
      if (mode == 2) v = geluf(v);
      s += v; ss += v * v;
    }
#pragma unroll
    for (int m = 16; m >= 1; m >>= 1) { s += __shfl_xor(s, m, 32); ss += __shfl_xor(ss, m, 32); }
    float mean = s / N;
    float var  = ss / N - mean * mean;
    float inv  = rsqrtf(var + 1e-5f);
    for (int c = lane; c < N; c += 32) {
      float v = tile[row * N + c];
      if (mode == 2) v = geluf(v);
      out[(size_t)(rt * 16 + row) * ldo + colofs + c] = (v - mean) * inv * g[c] + be[c];
    }
  }
}

// ===========================================================================
// 4 LSTM scans (eye fwd/bwd, flight fwd/bwd), one persistent block each.
// gates[32,256] = x@Wih^T (tiny K<=5, scalar) + h@Whh^T (K=64, WMMA) + biases.
// h state kept in LDS as f16 for A-frag reuse. grid.x = 4, blockDim = 256.
// ===========================================================================
__global__ void lstm_scan(const float* __restrict__ xe, const float* __restrict__ xf,
                          const float* wih_ef, const float* bih_ef, const float* bhh_ef,
                          const float* wih_eb, const float* bih_eb, const float* bhh_eb,
                          const float* wih_ff, const float* bih_ff, const float* bhh_ff,
                          const float* wih_fb, const float* bih_fb, const float* bhh_fb,
                          const _Float16* __restrict__ whh16,
                          float* __restrict__ hcat_eye, float* __restrict__ hcat_fl) {
  __shared__ _Float16 h16[32 * 64];
  __shared__ float    gbuf[32 * 256];
  __shared__ float    cbuf[32 * 64];
  __shared__ float    wihc[256 * 5];
  __shared__ float    bsum[256];

  const int dir  = blockIdx.x;
  const int tid  = threadIdx.x;
  const int lane = tid & 31;
  const int w    = tid >> 5;
  const int half8 = (lane < 16) ? 0 : 8;
  const int mrow  = lane & 15;

  const float* x; const float* wih; const float* bih; const float* bhh;
  float* outp; int ind; int colofs; int rev;
  if (dir == 0)      { x = xe; wih = wih_ef; bih = bih_ef; bhh = bhh_ef; outp = hcat_eye; ind = 5; colofs = 0;  rev = 0; }
  else if (dir == 1) { x = xe; wih = wih_eb; bih = bih_eb; bhh = bhh_eb; outp = hcat_eye; ind = 5; colofs = 64; rev = 1; }
  else if (dir == 2) { x = xf; wih = wih_ff; bih = bih_ff; bhh = bhh_ff; outp = hcat_fl;  ind = 3; colofs = 0;  rev = 0; }
  else               { x = xf; wih = wih_fb; bih = bih_fb; bhh = bhh_fb; outp = hcat_fl;  ind = 3; colofs = 64; rev = 1; }

  // Preload B-fragments of Whh^T (2 coltiles per wave x 2 ktiles) into registers.
  v16h bfr[2][2];
#pragma unroll
  for (int ct = 0; ct < 2; ++ct) {
    int n = w * 2 + ct;
#pragma unroll
    for (int kk = 0; kk < 2; ++kk)
      bfr[ct][kk] = load_v16h(whh16 + (size_t)(((dir * 2 + kk) * 16 + n) * 32 + lane) * 16);
  }

  for (int i = tid; i < 32 * 64; i += 256) { h16[i] = (_Float16)0.f; cbuf[i] = 0.f; }
  for (int i = tid; i < 256; i += 256) bsum[i] = bih[i] + bhh[i];
  for (int i = tid; i < 256 * ind; i += 256) wihc[i] = wih[i];
  __syncthreads();

  for (int t = 0; t < TT; ++t) {
    const int tg = rev ? (TT - 1 - t) : t;

    // gates tile via WMMA: h[32,64] @ Whh^T[64,256]
    v16h af[2][2];
#pragma unroll
    for (int m = 0; m < 2; ++m)
#pragma unroll
      for (int kk = 0; kk < 2; ++kk) {
#pragma unroll
        for (int hh = 0; hh < 8; ++hh) {
          int k = kk * 32 + half8 + hh;
          af[m][kk][hh]     = h16[(m * 16 + mrow) * 64 + k];
          af[m][kk][hh + 8] = h16[(m * 16 + mrow) * 64 + k + 16];
        }
      }
#pragma unroll
    for (int ct = 0; ct < 2; ++ct)
#pragma unroll
      for (int m = 0; m < 2; ++m) {
        v8f acc = {0.f, 0.f, 0.f, 0.f, 0.f, 0.f, 0.f, 0.f};
#pragma unroll
        for (int kk = 0; kk < 2; ++kk) acc = WMMA(af[m][kk], bfr[ct][kk], acc);
        int col = (w * 2 + ct) * 16 + mrow;
#pragma unroll
        for (int r = 0; r < 8; ++r) gbuf[(m * 16 + r + half8) * 256 + col] = acc[r];
      }
    __syncthreads();

    // elementwise LSTM cell update; gate order i,f,g,o
#pragma unroll
    for (int ii = 0; ii < 8; ++ii) {
      int idx = (ii << 8) + tid;      // idx = b*64 + j
      int b = idx >> 6, j = idx & 63;
      float xv[5];
      for (int k = 0; k < ind; ++k) xv[k] = x[(size_t)(b * TT + tg) * ind + k];
      float g4[4];
#pragma unroll
      for (int gi = 0; gi < 4; ++gi) {
        int col = gi * 64 + j;
        float s = gbuf[b * 256 + col] + bsum[col];
        for (int k = 0; k < ind; ++k) s += xv[k] * wihc[col * ind + k];
        g4[gi] = s;
      }
      float cc = cbuf[idx];
      float cn = sigf(g4[1]) * cc + sigf(g4[0]) * tanhf(g4[2]);
      float hn = sigf(g4[3]) * tanhf(cn);
      cbuf[idx] = cn;
      h16[idx]  = (_Float16)hn;
      outp[(size_t)(b * TT + tg) * 128 + colofs + j] = hn;
    }
    __syncthreads();
  }
}

// ===========================================================================
// Flash attention, one wave per (b, head, 16-query tile). D=32 => QK^T is a
// single WMMA per key tile; P@V is one A-frag (16x32 keys) x 2 B-frags.
// grid = B*H*(T/16) = 4096, blockDim = 32.
// ===========================================================================
__global__ void attn_kernel(const float* __restrict__ q, const float* __restrict__ k,
                            const float* __restrict__ v, float* __restrict__ o) {
  __shared__ _Float16 pbuf[16 * 32];
  const int lane  = threadIdx.x & 31;
  const int half8 = (lane < 16) ? 0 : 8;
  const int b16   = (lane < 16) ? 0 : 16;
  const int mrow  = lane & 15;
  const int qt = blockIdx.x & 31;
  const int hd = (blockIdx.x >> 5) & 3;
  const int b  = blockIdx.x >> 7;
  const float scale = 0.17677669529663687f;  // 1/sqrt(32)

  // Q fragment with softmax scale folded in.
  const float* qrow = q + (size_t)(b * TT + qt * 16 + mrow) * 128 + hd * 32;
  v16h aq;
#pragma unroll
  for (int hh = 0; hh < 8; ++hh) {
    aq[hh]     = (_Float16)(qrow[b16 + hh] * scale);
    aq[hh + 8] = (_Float16)(qrow[b16 + 8 + hh] * scale);
  }

  v8f o0 = {0.f, 0.f, 0.f, 0.f, 0.f, 0.f, 0.f, 0.f};
  v8f o1 = o0;
  float mr[8], lr[8];
#pragma unroll
  for (int r = 0; r < 8; ++r) { mr[r] = -3.0e38f; lr[r] = 0.f; }
  const v8f zero8 = {0.f, 0.f, 0.f, 0.f, 0.f, 0.f, 0.f, 0.f};

  for (int kc = 0; kc < TT / 32; ++kc) {
    // scores for 32 keys = two 16x16 WMMA tiles
    v8f s[2];
#pragma unroll
    for (int sub = 0; sub < 2; ++sub) {
      int key = kc * 32 + sub * 16 + mrow;
      const float* kb = k + (size_t)(b * TT + key) * 128 + hd * 32 + b16;
      v16h bk;
#pragma unroll
      for (int hh = 0; hh < 16; ++hh) bk[hh] = (_Float16)kb[hh];
      s[sub] = WMMA(aq, bk, zero8);
    }
    // online softmax (row spans 16 lanes of each half; shfl masks 1..8)
#pragma unroll
    for (int r = 0; r < 8; ++r) {
      float cm = fmaxf(s[0][r], s[1][r]);
#pragma unroll
      for (int m = 8; m >= 1; m >>= 1) cm = fmaxf(cm, __shfl_xor(cm, m, 32));
      float mn    = fmaxf(mr[r], cm);
      float alpha = __expf(mr[r] - mn);
      mr[r] = mn;
      float p0 = __expf(s[0][r] - mn);
      float p1 = __expf(s[1][r] - mn);
      float rs = p0 + p1;
#pragma unroll
      for (int m = 8; m >= 1; m >>= 1) rs += __shfl_xor(rs, m, 32);
      lr[r] = lr[r] * alpha + rs;
      o0[r] *= alpha; o1[r] *= alpha;
      pbuf[(r + half8) * 32 + mrow]      = (_Float16)p0;
      pbuf[(r + half8) * 32 + 16 + mrow] = (_Float16)p1;
    }
    __syncthreads();
    // P (16x32, f16) as A-frag via LDS transpose
    v16h pa;
#pragma unroll
    for (int hh = 0; hh < 8; ++hh) {
      pa[hh]     = pbuf[mrow * 32 + b16 + hh];
      pa[hh + 8] = pbuf[mrow * 32 + b16 + 8 + hh];
    }
#pragma unroll
    for (int dsub = 0; dsub < 2; ++dsub) {
      v16h bv;
#pragma unroll
      for (int hh = 0; hh < 16; ++hh)
        bv[hh] = (_Float16)v[(size_t)(b * TT + kc * 32 + b16 + hh) * 128 + hd * 32 + dsub * 16 + mrow];
      if (dsub == 0) o0 = WMMA(pa, bv, o0); else o1 = WMMA(pa, bv, o1);
    }
    __syncthreads();
  }

#pragma unroll
  for (int r = 0; r < 8; ++r) {
    float invl = 1.f / lr[r];
    size_t base = (size_t)(b * TT + qt * 16 + r + half8) * 128 + hd * 32;
    o[base + mrow]      = o0[r] * invl;
    o[base + 16 + mrow] = o1[r] * invl;
  }
}

// ===========================================================================
// ConvLSTM scan (1x1 spatial => depthwise = scale+bias). One persistent block.
// gates[32,512] = (comb*dw_w+dw_b)[32,256] @ pw_W^T via WMMA (K=256),
// gate order i,f,o,g; per-step LayerNorm(128) feeds back into h.
// c-state lives in registers (16 per thread). blockDim = 256.
// ===========================================================================
__global__ void convlstm_scan(const float* __restrict__ fused,
                              const float* __restrict__ dww, const float* __restrict__ dwb,
                              const _Float16* __restrict__ pw16, const float* __restrict__ pwb,
                              const float* __restrict__ lg, const float* __restrict__ lb,
                              float* __restrict__ hout) {
  __shared__ _Float16 gbuf[32 * 512];  // phase 1/2: comb f16 [32*256]; phase 3/4: gates f16
  __shared__ float    hb[32 * 128];
  const int tid  = threadIdx.x;
  const int lane = tid & 31;
  const int w    = tid >> 5;
  const int half8 = (lane < 16) ? 0 : 8;
  const int mrow  = lane & 15;
  const int bown = tid >> 3;          // row owned for cell update (8 threads/row)
  const int jb   = (tid & 7) * 16;

  float creg[16];
#pragma unroll
  for (int i = 0; i < 16; ++i) creg[i] = 0.f;
  for (int i = tid; i < 32 * 128; i += 256) hb[i] = 0.f;
  __syncthreads();

  for (int t = 0; t < TT; ++t) {
    // P1: comb = [x_t | h], depthwise center tap, store f16
#pragma unroll
    for (int i = 0; i < 32; ++i) {
      int col = (tid & 7) * 32 + i;
      int b   = tid >> 3;
      float vv = (col < 128) ? fused[(size_t)(b * TT + t) * 128 + col] : hb[b * 128 + (col - 128)];
      gbuf[b * 256 + col] = (_Float16)(vv * dww[col] + dwb[col]);
    }
    __syncthreads();

    // P2: WMMA gates: each wave 4 coltiles x 2 rowtiles x 8 ktiles
    v8f acc[4][2];
#pragma unroll
    for (int ct = 0; ct < 4; ++ct)
#pragma unroll
      for (int m = 0; m < 2; ++m) {
        v8f z = {0.f, 0.f, 0.f, 0.f, 0.f, 0.f, 0.f, 0.f};
        acc[ct][m] = z;
      }
    for (int kk = 0; kk < 8; ++kk) {
      v16h a0, a1;
#pragma unroll
      for (int hh = 0; hh < 8; ++hh) {
        int kA = kk * 32 + half8 + hh;
        a0[hh] = gbuf[(mrow)      * 256 + kA];  a0[hh + 8] = gbuf[(mrow)      * 256 + kA + 16];
        a1[hh] = gbuf[(16 + mrow) * 256 + kA];  a1[hh + 8] = gbuf[(16 + mrow) * 256 + kA + 16];
      }
#pragma unroll
      for (int ct = 0; ct < 4; ++ct) {
        int n = w * 4 + ct;
        v16h bf = load_v16h(pw16 + (size_t)((kk * 32 + n) * 32 + lane) * 16);
        acc[ct][0] = WMMA(a0, bf, acc[ct][0]);
        acc[ct][1] = WMMA(a1, bf, acc[ct][1]);
      }
    }
    __syncthreads();

    // P3: spill gates (f16) to LDS
#pragma unroll
    for (int ct = 0; ct < 4; ++ct) {
      int col = (w * 4 + ct) * 16 + mrow;
#pragma unroll
      for (int m = 0; m < 2; ++m)
#pragma unroll
        for (int r = 0; r < 8; ++r)
          gbuf[(m * 16 + r + half8) * 512 + col] = (_Float16)acc[ct][m][r];
    }
    __syncthreads();

    // P4: cell update (i,f,o,g order) + per-step LayerNorm over 128
    float hr[16], s = 0.f, ss = 0.f;
#pragma unroll
    for (int i = 0; i < 16; ++i) {
      int j = jb + i;
      float gi = (float)gbuf[bown * 512 + j]        + pwb[j];
      float gf = (float)gbuf[bown * 512 + 128 + j]  + pwb[128 + j];
      float go = (float)gbuf[bown * 512 + 256 + j]  + pwb[256 + j];
      float gg = (float)gbuf[bown * 512 + 384 + j]  + pwb[384 + j];
      creg[i] = sigf(gf) * creg[i] + sigf(gi) * tanhf(gg);
      float h = sigf(go) * tanhf(creg[i]);
      hr[i] = h; s += h; ss += h * h;
    }
#pragma unroll
    for (int m = 4; m >= 1; m >>= 1) { s += __shfl_xor(s, m, 32); ss += __shfl_xor(ss, m, 32); }
    float mean = s / 128.f;
    float var  = ss / 128.f - mean * mean;
    float inv  = rsqrtf(var + 1e-5f);
#pragma unroll
    for (int i = 0; i < 16; ++i) {
      int j = jb + i;
      hb[bown * 128 + j] = (hr[i] - mean) * inv * lg[j] + lb[j];
    }
    __syncthreads();
  }

#pragma unroll
  for (int i = 0; i < 16; ++i) hout[bown * 128 + jb + i] = hb[bown * 128 + jb + i];
}

// Final classifier: [32,64] @ [10,64]^T + b -> [32,10]
__global__ void head_final(const float* __restrict__ h, const float* __restrict__ W2,
                           const float* __restrict__ b2, float* __restrict__ out) {
  int tid = threadIdx.x;
  if (tid >= 320) return;
  int b = tid / 10, n = tid % 10;
  float s = b2[n];
  for (int k = 0; k < 64; ++k) s += h[b * 64 + k] * W2[n * 64 + k];
  out[b * 10 + n] = s;
}

// ===========================================================================
extern "C" void kernel_launch(void* const* d_in, const int* in_sizes, int n_in,
                              void* d_out, int out_size, void* d_ws, size_t ws_size,
                              hipStream_t stream) {
  (void)in_sizes; (void)n_in; (void)out_size; (void)ws_size;
#define P(i) ((const float*)d_in[(i)])
  const float* img = P(0);
  const float* xe  = P(1);
  const float* xf  = P(2);
  // params (flattened dict order): 3 imgW 4 imgb 5 img_g 6 img_be
  // 7..10 eye_f(Wih,Whh,bih,bhh) 11..14 eye_b 15..18 fl_f 19..22 fl_b
  // 23..26 eyeW,b,g,be  27..30 flW,...  31..34 caTxt  35..38 caImg
  // 39..42 Wq Wk Wv Wo  43..46 bq bk bv bo  47,48 an_g,an_be
  // 49,50 dw_w,dw_b  51,52 pw_W,pw_b  53,54 lnorm  55..58 cls1  59,60 cls2

  float* w = (float*)d_ws;
  float* bImg = w;                        // [BT,256] img_ln; later reused as attn out [BT,128]
  float* bHE  = bImg + (size_t)BT * 256;  // hcat_eye -> q
  float* bHF  = bHE + (size_t)BT * 128;   // hcat_fl  -> k
  float* bTXT = bHF + (size_t)BT * 128;   // txt      -> v
  float* bQ   = bTXT + (size_t)BT * 128;  // Q_ln     -> fused
  float* bK   = bQ + (size_t)BT * 128;    // K_ln
  float* bHfin = bK + (size_t)BT * 128;   // [32,128]
  float* bHead = bHfin + 32 * 128;        // [32,64]
  _Float16* whh16 = (_Float16*)(bHead + 32 * 64);
  _Float16* pw16  = whh16 + 65536;

  // 0) pre-swizzle recurrent weights into WMMA B-frag layout
  pack_frags<<<768, 256, 0, stream>>>(P(8), P(12), P(16), P(20), P(51), whh16, pw16);

  // 1) img projection + LN  [BT,1280] -> [BT,256]
  gemm_wmma<<<1024, 512, 0, stream>>>(img, 1280, P(3), P(4), P(5), P(6), bImg, 256, 0, 1);

  // 2) four BiLSTM scans
  lstm_scan<<<4, 256, 0, stream>>>(xe, xf,
                                   P(7), P(9), P(10),  P(11), P(13), P(14),
                                   P(15), P(17), P(18), P(19), P(21), P(22),
                                   whh16, bHE, bHF);

  // 3) eye / fl projections + LN -> concat into txt[BT,128]
  gemm_wmma<<<1024, 128, 0, stream>>>(bHE, 128, P(23), P(24), P(25), P(26), bTXT, 128, 0,  1);
  gemm_wmma<<<1024, 128, 0, stream>>>(bHF, 128, P(27), P(28), P(29), P(30), bTXT, 128, 64, 1);

  // 4) cross-attention pre-projections + LN
  gemm_wmma<<<1024, 256, 0, stream>>>(bTXT, 128, P(31), P(32), P(33), P(34), bQ, 128, 0, 1);
  gemm_wmma<<<1024, 256, 0, stream>>>(bImg, 256, P(35), P(36), P(37), P(38), bK, 128, 0, 1);

  // 5) q/k/v projections (bias only)
  gemm_wmma<<<1024, 256, 0, stream>>>(bQ, 128, P(39), P(43), nullptr, nullptr, bHE,  128, 0, 0);
  gemm_wmma<<<1024, 256, 0, stream>>>(bK, 128, P(40), P(44), nullptr, nullptr, bHF,  128, 0, 0);
  gemm_wmma<<<1024, 256, 0, stream>>>(bK, 128, P(41), P(45), nullptr, nullptr, bTXT, 128, 0, 0);

  // 6) flash attention -> bImg (reused)
  attn_kernel<<<4096, 32, 0, stream>>>(bHE, bHF, bTXT, bImg);

  // 7) output projection + LN -> fused
  gemm_wmma<<<1024, 256, 0, stream>>>(bImg, 128, P(42), P(46), P(47), P(48), bQ, 128, 0, 1);

  // 8) ConvLSTM scan -> final hidden [32,128]
  convlstm_scan<<<1, 256, 0, stream>>>(bQ, P(49), P(50), pw16, P(52), P(53), P(54), bHfin);

  // 9) classifier head: GEMM + GELU + LN, then final linear
  gemm_wmma<<<2, 128, 0, stream>>>(bHfin, 128, P(55), P(56), P(57), P(58), bHead, 64, 0, 2);
  head_final<<<1, 320, 0, stream>>>(bHead, P(59), P(60), (float*)d_out);
#undef P
}
